// DiffusionModule_predict_X0_74698071212335
// MI455X (gfx1250) — matrix operand order repaired and checked by reading, not compiled, not run on hardware
//
#include <hip/hip_runtime.h>
#include <hip/hip_bf16.h>

// ---------------------------------------------------------------------------
// CDNA5 / gfx1250 (MI455X) implementation.
// wave32, WMMA v_wmma_f32_16x16x32_bf16 for all GEMMs, flash-style masked
// attention (32 keys/iter, bit-packed mask, scale folded into Wq), DAD folded
// into WMMA epilogues, A converted once (vectorized) to bf16 so all O(N^2)
// passes are L2-resident (32MB << 192MB L2).
// ---------------------------------------------------------------------------

#define NN 4096
#define DD 256
#define FF 64
#define PP 16

typedef __bf16 bf16x16 __attribute__((ext_vector_type(16)));
typedef float  f32x8   __attribute__((ext_vector_type(8)));
typedef float  f32x4   __attribute__((ext_vector_type(4)));
typedef unsigned int u32x4 __attribute__((ext_vector_type(4)));
typedef unsigned short u16x4 __attribute__((ext_vector_type(4)));

union V16 { bf16x16 v; u32x4 q[2]; unsigned short s[16]; };

__device__ __forceinline__ unsigned short f2bf(float f) {
  unsigned u = __float_as_uint(f);
  u += 0x7FFFu + ((u >> 16) & 1u);          // round-to-nearest-even
  return (unsigned short)(u >> 16);
}
__device__ __forceinline__ unsigned short f2bf_trunc(float f) {
  return (unsigned short)(__float_as_uint(f) >> 16);
}

// A-operand (16xK tile row): lane-half h reads contiguous k = [kb .. kb+15]
template <int K>
__device__ __forceinline__ bf16x16 load_a_op(const unsigned short* rowptr, int kb) {
  V16 t;
  if (K % 32 == 0) {                         // constant-folds: unconditional b128 pair
    t.q[0] = *(const u32x4*)(rowptr + kb);
    t.q[1] = *(const u32x4*)(rowptr + kb + 8);
  } else {
    u32x4 z = {0, 0, 0, 0};
    t.q[0] = (kb + 8  <= K) ? *(const u32x4*)(rowptr + kb)      : z;
    t.q[1] = (kb + 16 <= K) ? *(const u32x4*)(rowptr + kb + 8)  : z;
  }
  return t.v;
}
// B-operand (Kx16 tile, stored transposed [n][k]): lane-half h reads contiguous k = [kh .. kh+15]
template <int K>
__device__ __forceinline__ bf16x16 load_b_op(const unsigned short* rowptr, int kh) {
  V16 t;
  if (K % 32 == 0) {
    t.q[0] = *(const u32x4*)(rowptr + kh);
    t.q[1] = *(const u32x4*)(rowptr + kh + 8);
  } else {
    u32x4 z = {0, 0, 0, 0};
    t.q[0] = (kh + 8  <= K) ? *(const u32x4*)(rowptr + kh)     : z;
    t.q[1] = (kh + 16 <= K) ? *(const u32x4*)(rowptr + kh + 8) : z;
  }
  return t.v;
}

// ------------------------------ elementwise --------------------------------

// vectorized f32 -> bf16 (n4 = count/4); the only HBM-bound pass runs at b128
__global__ void k_f2bf(const f32x4* __restrict__ in, u16x4* __restrict__ out, size_t n4) {
  size_t i = (size_t)blockIdx.x * blockDim.x + threadIdx.x;
  size_t stride = (size_t)gridDim.x * blockDim.x;
  for (; i < n4; i += stride) {
    f32x4 x = in[i];
    u16x4 o;
#pragma unroll
    for (int j = 0; j < 4; ++j) o[j] = f2bf(x[j]);
    out[i] = o;
  }
}

// Wt[c][r] = bf16(scale * W[r][c])   (make [Nout][K] layout for B-operands)
__global__ void k_tconv(const float* __restrict__ W, unsigned short* __restrict__ Wt,
                        int R, int Cc, float scale) {
  int i = blockIdx.x * blockDim.x + threadIdx.x;
  if (i >= R * Cc) return;
  int r = i / Cc, c = i - r * Cc;
  Wt[(size_t)c * R + r] = f2bf(scale * W[(size_t)r * Cc + c]);
}

// bit-packed transposed adjacency(+self-loop) mask: bit j of Mt[n][mb] = (A_hat[mb*8+j][n] > 0)
__global__ void k_maskbits(const unsigned short* __restrict__ Ab, unsigned char* __restrict__ Mt) {
  int i = blockIdx.x * blockDim.x + threadIdx.x;
  if (i >= NN * (NN / 8)) return;
  int n = i & (NN - 1);
  int mb = i >> 12;                         // NN == 4096
  unsigned bits = 0;
#pragma unroll
  for (int j = 0; j < 8; ++j) {
    int m = mb * 8 + j;
    unsigned short au = Ab[(size_t)m * NN + n];
    bool pos = (au != 0) && ((au & 0x8000u) == 0);
    bits |= (unsigned)(pos || (m == n)) << j;
  }
  Mt[(size_t)n * (NN / 8) + mb] = (unsigned char)bits;
}

// deg, d = deg^-1/2 (inf->0), pro2 = X[:,P-2], Xdt[n][j] = bf16(d_j * X[j][n])
__global__ void k_rowstats(const float* __restrict__ A, const float* __restrict__ X,
                           float* __restrict__ deg, float* __restrict__ dinv,
                           float* __restrict__ pro2, unsigned short* __restrict__ Xdt) {
  int row = blockIdx.x;
  __shared__ float red[256];
  __shared__ float dsh;
  const float* ar = A + (size_t)row * NN;
  float s = 0.f;
  for (int c = threadIdx.x; c < NN; c += 256) s += ar[c];
  red[threadIdx.x] = s;
  __syncthreads();
  for (int st = 128; st; st >>= 1) {
    if ((int)threadIdx.x < st) red[threadIdx.x] += red[threadIdx.x + st];
    __syncthreads();
  }
  if (threadIdx.x == 0) {
    float dg = red[0];
    deg[row] = dg;
    float dv = dg > 0.f ? rsqrtf(dg) : 0.f;
    dinv[row] = dv; dsh = dv;
    pro2[row] = X[(size_t)row * PP + (PP - 2)];
  }
  __syncthreads();
  if ((int)threadIdx.x < PP)
    Xdt[(size_t)threadIdx.x * NN + row] = f2bf(dsh * X[(size_t)row * PP + threadIdx.x]);
}

// x_t_hidden = LN(x_t[i] * mlp_w + mlp_b) over D=256; one wave per row
__global__ void k_xth(const float* __restrict__ x_t, const float* __restrict__ mw,
                      const float* __restrict__ mb, const float* __restrict__ g,
                      const float* __restrict__ b, float* __restrict__ out, int M) {
  int row = blockIdx.x * blockDim.y + threadIdx.y;
  if (row >= M) return;
  int lane = threadIdx.x;
  float xv = x_t[row];
  float x[8]; float s = 0.f;
#pragma unroll
  for (int j = 0; j < 8; ++j) { int c = lane + 32 * j; x[j] = xv * mw[c] + mb[c]; s += x[j]; }
#pragma unroll
  for (int o = 16; o; o >>= 1) s += __shfl_xor(s, o, 32);
  float mean = s * (1.f / 256.f);
  float vs = 0.f;
#pragma unroll
  for (int j = 0; j < 8; ++j) { float d2 = x[j] - mean; vs += d2 * d2; }
#pragma unroll
  for (int o = 16; o; o >>= 1) vs += __shfl_xor(vs, o, 32);
  float inv = rsqrtf(vs * (1.f / 256.f) + 1e-5f);
#pragma unroll
  for (int j = 0; j < 8; ++j) {
    int c = lane + 32 * j;
    out[(size_t)row * 256 + c] = (x[j] - mean) * inv * g[c] + b[c];
  }
}

// generic row-LN (+optional relu, +optional per-row scalar add), writes f32 and/or bf16
template <int C>
__global__ void k_layernorm(const float* __restrict__ in, const float* __restrict__ g,
                            const float* __restrict__ b, int relu,
                            const float* __restrict__ rowadd,
                            float* __restrict__ outf, unsigned short* __restrict__ outb, int M) {
  int row = blockIdx.x * blockDim.y + threadIdx.y;
  if (row >= M) return;
  int lane = threadIdx.x;
  const float* rp = in + (size_t)row * C;
  const int nj = C / 32;
  float x[C / 32]; float s = 0.f;
#pragma unroll
  for (int j = 0; j < nj; ++j) { x[j] = rp[lane + 32 * j]; s += x[j]; }
#pragma unroll
  for (int o = 16; o; o >>= 1) s += __shfl_xor(s, o, 32);
  float mean = s * (1.f / C);
  float vs = 0.f;
#pragma unroll
  for (int j = 0; j < nj; ++j) { float d2 = x[j] - mean; vs += d2 * d2; }
#pragma unroll
  for (int o = 16; o; o >>= 1) vs += __shfl_xor(vs, o, 32);
  float inv = rsqrtf(vs * (1.f / C) + 1e-5f);
  float ra = rowadd ? rowadd[row] : 0.f;
#pragma unroll
  for (int j = 0; j < nj; ++j) {
    int c = lane + 32 * j;
    float y = (x[j] - mean) * inv * g[c] + b[c];
    if (relu) y = fmaxf(y, 0.f);
    y += ra;
    if (outf) outf[(size_t)row * C + c] = y;
    if (outb) outb[(size_t)row * C + c] = f2bf(y);
  }
}

// enc_in = x_t_hidden + pos_em + cross ; Zt[n][j] = bf16(d_j * enc_in[j][n])
__global__ void k_encin(const float* __restrict__ xth, const float* __restrict__ cross,
                        const int* __restrict__ tptr, const float* __restrict__ dinv,
                        float* __restrict__ encin, unsigned short* __restrict__ Zt) {
  int row = blockIdx.x, c = threadIdx.x;        // blockDim.x == 256
  float t = (float)tptr[0];
  int i2 = c >> 1;
  float ang = t * __expf((float)(2 * i2) * (-9.2103403719761836f / 256.f));
  float pe = (c & 1) ? __cosf(ang) : __sinf(ang);
  float v = xth[(size_t)row * DD + c] + pe + cross[(size_t)row * DD + c];
  encin[(size_t)row * DD + c] = v;
  Zt[(size_t)c * NN + row] = f2bf(dinv[row] * v);
}

// out[i] = dot([enc_i, enh_i], w) + b + pro2[i]
__global__ void k_final(const float* __restrict__ enc, const float* __restrict__ enh,
                        const float* __restrict__ w, const float* __restrict__ bptr,
                        const float* __restrict__ pro2, float* __restrict__ out, int M) {
  int row = blockIdx.x * blockDim.y + threadIdx.y;
  if (row >= M) return;
  int lane = threadIdx.x;
  float s = 0.f;
#pragma unroll
  for (int j = 0; j < 4; ++j) {
    int c = lane + 32 * j;
    float x = (c < 64) ? enc[(size_t)row * 64 + c] : enh[(size_t)row * 64 + (c - 64)];
    s += x * w[c];
  }
#pragma unroll
  for (int o = 16; o; o >>= 1) s += __shfl_xor(s, o, 32);
  if (lane == 0) out[row] = s + bptr[0] + pro2[row];
}

// ------------------------------ WMMA GEMMs ---------------------------------

// C[M,Nn] = A[M,K](bf16 row-major) @ Bt[Nn,K]^T (bf16) + bias.
// One wave computes a 16 x (16*NT) tile: A-fragment reused across NT WMMAs.
template <int K, int NT>
__global__ void __launch_bounds__(32) k_gemm_bt(
    const unsigned short* __restrict__ A, int lda,
    const unsigned short* __restrict__ Bt, int ldb,
    const float* __restrict__ bias,
    float* __restrict__ C, unsigned short* __restrict__ Cb, int ldc) {
  const int lane = threadIdx.x & 31;
  const int r = lane & 15, half = lane >> 4;
  const int n0 = (blockIdx.x * NT) << 4, m0 = blockIdx.y << 4;
  const unsigned short* Arow = A + (size_t)(m0 + r) * lda;
  const unsigned short* Brow[NT];
#pragma unroll
  for (int t = 0; t < NT; ++t) Brow[t] = Bt + (size_t)(n0 + t * 16 + r) * ldb;
  f32x8 acc[NT];
#pragma unroll
  for (int t = 0; t < NT; ++t)
#pragma unroll
    for (int v = 0; v < 8; ++v) acc[t][v] = 0.f;

#pragma unroll 2
  for (int k0 = 0; k0 < K; k0 += 32) {
    bf16x16 a = load_a_op<K>(Arow, k0 + half * 16);
#pragma unroll
    for (int t = 0; t < NT; ++t) {
      bf16x16 b = load_b_op<K>(Brow[t], k0 + half * 16);
      acc[t] = __builtin_amdgcn_wmma_f32_16x16x32_bf16(false, a, false, b, (short)0, acc[t], false, false);
    }
  }
#pragma unroll
  for (int t = 0; t < NT; ++t) {
    float bn = bias ? bias[n0 + t * 16 + r] : 0.f;
#pragma unroll
    for (int v = 0; v < 8; ++v) {
      int m = m0 + half * 8 + v;
      float val = acc[t][v] + bn;
      if (C)  C[(size_t)m * ldc + n0 + t * 16 + r] = val;
      if (Cb) Cb[(size_t)m * ldc + n0 + t * 16 + r] = f2bf(val);
    }
  }
}

// Out = DAD @ Y without materializing DAD:  out_i = d_i * (A @ (d .* Y))_i + d_i^2 * Y_i
template <int NT>
__global__ void __launch_bounds__(32) k_dad_gemm(
    const unsigned short* __restrict__ Ab,
    const unsigned short* __restrict__ Bt,
    const float* __restrict__ dinv,
    const float* __restrict__ Y, int ldy,
    unsigned short* __restrict__ Outb) {
  const int lane = threadIdx.x & 31;
  const int r = lane & 15, half = lane >> 4;
  const int n0 = (blockIdx.x * NT) << 4, m0 = blockIdx.y << 4;
  const unsigned short* Arow = Ab + (size_t)(m0 + r) * NN;
  const unsigned short* Brow[NT];
#pragma unroll
  for (int t = 0; t < NT; ++t) Brow[t] = Bt + (size_t)(n0 + t * 16 + r) * (size_t)NN;
  f32x8 acc[NT];
#pragma unroll
  for (int t = 0; t < NT; ++t)
#pragma unroll
    for (int v = 0; v < 8; ++v) acc[t][v] = 0.f;

#pragma unroll 2
  for (int k0 = 0; k0 < NN; k0 += 32) {
    if (k0 + 512 < NN) __builtin_prefetch(Arow + k0 + 512, 0, 1);   // global_prefetch ahead in L2
    bf16x16 a = load_a_op<NN>(Arow, k0 + half * 16);
#pragma unroll
    for (int t = 0; t < NT; ++t) {
      bf16x16 b = load_b_op<NN>(Brow[t], k0 + half * 16);
      acc[t] = __builtin_amdgcn_wmma_f32_16x16x32_bf16(false, a, false, b, (short)0, acc[t], false, false);
    }
  }
#pragma unroll
  for (int t = 0; t < NT; ++t)
#pragma unroll
    for (int v = 0; v < 8; ++v) {
      int m = m0 + half * 8 + v;
      float di = dinv[m];
      float val = di * acc[t][v] + di * di * Y[(size_t)m * ldy + n0 + t * 16 + r];
      Outb[(size_t)m * ldy + n0 + t * 16 + r] = f2bf(val);
    }
}

// Flash-style adjacency-masked MHA: one wave = (head, 16 query rows), 32 keys/iter,
// online softmax, bit-packed mask (1 u8 load covers 8 query rows), branch-free loads.
// 1/sqrt(dh) is pre-folded into Wq, so scores come out of the WMMA already scaled.
__global__ void __launch_bounds__(32) k_flash(
    const unsigned short* __restrict__ Qb, const unsigned short* __restrict__ Kb,
    const unsigned short* __restrict__ Vt, const unsigned char* __restrict__ Mt,
    unsigned short* __restrict__ Ob) {
  __shared__ __align__(16) unsigned short pst[16 * 32];
  const int lane = threadIdx.x & 31;
  const int r = lane & 15, half = lane >> 4;
  const int h = blockIdx.y;
  const int q0 = blockIdx.x << 4;
  const u32x4 z = {0, 0, 0, 0};
  const unsigned hmv = (half == 0) ? 0xFFFFFFFFu : 0u;    // zero pad for QK B-operand half1
  const u32x4 hm = {hmv, hmv, hmv, hmv};

  // Q as A-operand once (dh = 16, zero-padded to K = 32)
  V16 qa;
  qa.q[0] = *(const u32x4*)(Qb + (size_t)(q0 + r) * FF + h * 16) & hm;
  qa.q[1] = *(const u32x4*)(Qb + (size_t)(q0 + r) * FF + h * 16 + 8) & hm;

  float rowmax[8], rowsum[8];
  f32x8 acc;
#pragma unroll
  for (int v = 0; v < 8; ++v) { rowmax[v] = -1e30f; rowsum[v] = 0.f; acc[v] = 0.f; }
  const int mrow = q0 + half * 8;
  const int mbyte = (q0 >> 3) + half;       // Mt byte covering rows mrow..mrow+7

  for (int kb0 = 0; kb0 < NN; kb0 += 32) {
    // two QK score tiles (keys kb0..+15 and kb0+16..+31); AND-mask pads half1 (no branches)
    const unsigned short* kp0 = Kb + (size_t)(kb0 + r) * FF + h * 16;
    const unsigned short* kp1 = kp0 + 16 * FF;
    V16 kB0, kB1;
    kB0.q[0] = *(const u32x4*)(kp0) & hm;
    kB0.q[1] = *(const u32x4*)(kp0 + 8) & hm;
    kB1.q[0] = *(const u32x4*)(kp1) & hm;
    kB1.q[1] = *(const u32x4*)(kp1 + 8) & hm;

    f32x8 s0, s1;
#pragma unroll
    for (int v = 0; v < 8; ++v) { s0[v] = 0.f; s1[v] = 0.f; }
    s0 = __builtin_amdgcn_wmma_f32_16x16x32_bf16(false, qa.v, false, kB0.v, (short)0, s0, false, false);
    s1 = __builtin_amdgcn_wmma_f32_16x16x32_bf16(false, qa.v, false, kB1.v, (short)0, s1, false, false);

    // one mask byte per key column covers all 8 rows of this lane
    unsigned m0b = Mt[(size_t)(kb0 + r) * (NN / 8) + mbyte];
    unsigned m1b = Mt[(size_t)(kb0 + 16 + r) * (NN / 8) + mbyte];

    float corr[8];
#pragma unroll
    for (int v = 0; v < 8; ++v) {
      float sv0 = ((m0b >> v) & 1u) ? s0[v] : -1e9f;   // scale pre-folded into Wq
      float sv1 = ((m1b >> v) & 1u) ? s1[v] : -1e9f;
      float mx = fmaxf(sv0, sv1);
#pragma unroll
      for (int o = 8; o; o >>= 1) mx = fmaxf(mx, __shfl_xor(mx, o, 16));
      float nm = fmaxf(rowmax[v], mx);
      corr[v] = __expf(rowmax[v] - nm);
      rowmax[v] = nm;
      float p0 = __expf(sv0 - nm);
      float p1 = __expf(sv1 - nm);
      float ps = p0 + p1;
#pragma unroll
      for (int o = 8; o; o >>= 1) ps += __shfl_xor(ps, o, 16);
      rowsum[v] = rowsum[v] * corr[v] + ps;
      pst[(half * 8 + v) * 32 + r]      = f2bf_trunc(p0);
      pst[(half * 8 + v) * 32 + 16 + r] = f2bf_trunc(p1);
    }

    // P tile (16x32, fully dense) reloaded in A-operand layout
    asm volatile("s_wait_dscnt 0" ::: "memory");
    V16 pa;
    pa.q[0] = *(const u32x4*)(&pst[r * 32 + half * 8]);
    pa.q[1] = *(const u32x4*)(&pst[r * 32 + 16 + half * 8]);

    // V as B-operand from Vt [FF][NN]: 32 real keys -> both halves dense, branch-free
    const unsigned short* vp = Vt + (size_t)(h * 16 + r) * NN + kb0 + half * 8;
    V16 vB;
    vB.q[0] = *(const u32x4*)(vp);
    vB.q[1] = *(const u32x4*)(vp + 16);

#pragma unroll
    for (int v = 0; v < 8; ++v) acc[v] *= corr[v];
    acc = __builtin_amdgcn_wmma_f32_16x16x32_bf16(false, pa.v, false, vB.v, (short)0, acc, false, false);
  }

#pragma unroll
  for (int v = 0; v < 8; ++v)
    Ob[(size_t)(mrow + v) * FF + h * 16 + r] = f2bf(acc[v] / rowsum[v]);
}

// ------------------------------ host side ----------------------------------

extern "C" void kernel_launch(void* const* d_in, const int* in_sizes, int n_in,
                              void* d_out, int out_size, void* d_ws, size_t ws_size,
                              hipStream_t stream) {
  (void)in_sizes; (void)n_in; (void)out_size; (void)ws_size;
  const int N = NN, D = DD, F = FF, P = PP;

  // inputs (setup_inputs order, params dict flattened in insertion order)
  const float* x_t   = (const float*)d_in[0];
  const int*   ts    = (const int*)d_in[1];
  const float* pro   = (const float*)d_in[2];
  const float* Afull = (const float*)d_in[3];
  const float* feat  = (const float*)d_in[4];
  const float* mlp_w = (const float*)d_in[5];
  const float* mlp_b = (const float*)d_in[6];
  const float* n1g   = (const float*)d_in[7];
  const float* n1b   = (const float*)d_in[8];
  // 9..16: high path (dead code, eliminated)
  const float* low_f1 = (const float*)d_in[17];
  const float* low_b1 = (const float*)d_in[18];
  const float* low_f2 = (const float*)d_in[19];
  const float* low_b2 = (const float*)d_in[20];
  const float* ln1g   = (const float*)d_in[21];
  const float* ln1b   = (const float*)d_in[22];
  const float* ln2g   = (const float*)d_in[23];
  const float* ln2b   = (const float*)d_in[24];
  const float* att_wv = (const float*)d_in[25];
  const float* att_bv = (const float*)d_in[26];
  const float* att_wo = (const float*)d_in[27];
  const float* att_bo = (const float*)d_in[28];
  const float* cng    = (const float*)d_in[29];
  const float* cnb    = (const float*)d_in[30];
  const float* de_conv = (const float*)d_in[31];
  const float* de_bias = (const float*)d_in[32];
  const float* dng     = (const float*)d_in[33];
  const float* dnb     = (const float*)d_in[34];
  const float* de_mlp_w = (const float*)d_in[35];
  const float* de_mlp_b = (const float*)d_in[36];
  const float* wq = (const float*)d_in[37];
  const float* wk = (const float*)d_in[38];
  const float* wvg = (const float*)d_in[39];
  const float* wog = (const float*)d_in[40];

  // workspace arena
  char* w = (char*)d_ws;
  size_t off = 0;
  auto arena = [&](size_t bytes) -> void* {
    void* p = w + off;
    off += (bytes + 255) & ~(size_t)255;
    return p;
  };
  unsigned short* Ab   = (unsigned short*)arena((size_t)N * N * 2);
  unsigned char*  Mt   = (unsigned char*)arena((size_t)N * (N / 8));
  float* deg  = (float*)arena((size_t)N * 4);
  float* dinv = (float*)arena((size_t)N * 4);
  float* pro2 = (float*)arena((size_t)N * 4);
  unsigned short* Xdt    = (unsigned short*)arena((size_t)P * N * 2);
  unsigned short* lowraw = (unsigned short*)arena((size_t)N * P * 2);
  float* xth  = (float*)arena((size_t)N * D * 4);
  float* bufA = (float*)arena((size_t)N * D * 4);
  float* bufB = (float*)arena((size_t)N * D * 4);
  unsigned short* bbA = (unsigned short*)arena((size_t)N * D * 2);
  unsigned short* bbB = (unsigned short*)arena((size_t)N * D * 2);
  float* encin = (float*)arena((size_t)N * D * 4);
  unsigned short* Zt = (unsigned short*)arena((size_t)D * N * 2);
  unsigned short* Yb = (unsigned short*)arena((size_t)N * D * 2);
  float* encf = (float*)arena((size_t)N * F * 4);
  unsigned short* featb = (unsigned short*)arena((size_t)N * F * 2);
  unsigned short* Qb  = (unsigned short*)arena((size_t)N * F * 2);
  unsigned short* Kbm = (unsigned short*)arena((size_t)N * F * 2);
  float* Vf = (float*)arena((size_t)N * F * 4);
  unsigned short* Vt = (unsigned short*)arena((size_t)F * N * 2);
  unsigned short* Ob = (unsigned short*)arena((size_t)N * F * 2);
  float* enh = (float*)arena((size_t)N * F * 4);
  unsigned short* lf1t = (unsigned short*)arena((size_t)128 * 16 * 2);
  unsigned short* lf2t = (unsigned short*)arena((size_t)256 * 128 * 2);
  unsigned short* wvb  = (unsigned short*)arena((size_t)D * D * 2);
  unsigned short* wob  = (unsigned short*)arena((size_t)D * D * 2);
  unsigned short* dct  = (unsigned short*)arena((size_t)F * D * 2);
  unsigned short* wqt  = (unsigned short*)arena((size_t)F * F * 2);
  unsigned short* wkt  = (unsigned short*)arena((size_t)F * F * 2);
  unsigned short* wvt  = (unsigned short*)arena((size_t)F * F * 2);
  unsigned short* wot  = (unsigned short*)arena((size_t)F * F * 2);

  const dim3 w8(32, 8);

  // stage: convert/transposes (one HBM pass over A; everything after hits L2)
  k_f2bf<<<2048, 256, 0, stream>>>((const f32x4*)Afull, (u16x4*)Ab, (size_t)N * N / 4);
  k_maskbits<<<(N * (N / 8)) / 256, 256, 0, stream>>>(Ab, Mt);
  k_rowstats<<<N, 256, 0, stream>>>(Afull, pro, deg, dinv, pro2, Xdt);
  k_xth<<<N / 8, w8, 0, stream>>>(x_t, mlp_w, mlp_b, n1g, n1b, xth, N);
  k_tconv<<<(16 * 128 + 255) / 256, 256, 0, stream>>>(low_f1, lf1t, 16, 128, 1.f);
  k_tconv<<<(128 * 256 + 255) / 256, 256, 0, stream>>>(low_f2, lf2t, 128, 256, 1.f);
  k_f2bf<<<64, 256, 0, stream>>>((const f32x4*)att_wv, (u16x4*)wvb, (size_t)D * D / 4);
  k_f2bf<<<64, 256, 0, stream>>>((const f32x4*)att_wo, (u16x4*)wob, (size_t)D * D / 4);
  k_tconv<<<(256 * 64 + 255) / 256, 256, 0, stream>>>(de_conv, dct, 256, 64, 1.f);
  k_tconv<<<16, 256, 0, stream>>>(wq, wqt, 64, 64, 0.25f);   // fold 1/sqrt(dh) into Wq
  k_tconv<<<16, 256, 0, stream>>>(wk, wkt, 64, 64, 1.f);
  k_tconv<<<16, 256, 0, stream>>>(wvg, wvt, 64, 64, 1.f);
  k_tconv<<<16, 256, 0, stream>>>(wog, wot, 64, 64, 1.f);
  k_f2bf<<<256, 256, 0, stream>>>((const f32x4*)feat, (u16x4*)featb, (size_t)N * F / 4);

  // low path: low_raw = DAD @ pro_dyn  (folded epilogue, K = 4096)
  k_dad_gemm<1><<<dim3(1, N / 16), 32, 0, stream>>>(Ab, Xdt, dinv, pro, P, lowraw);
  k_gemm_bt<16, 4><<<dim3(2, N / 16), 32, 0, stream>>>(lowraw, P, lf1t, 16, low_b1, bufA, nullptr, 128);
  k_layernorm<128><<<N / 8, w8, 0, stream>>>(bufA, ln1g, ln1b, 1, nullptr, nullptr, bbA, N);
  k_gemm_bt<128, 4><<<dim3(4, N / 16), 32, 0, stream>>>(bbA, 128, lf2t, 128, low_b2, bufB, nullptr, 256);
  k_layernorm<256><<<N / 8, w8, 0, stream>>>(bufB, ln2g, ln2b, 1, nullptr, nullptr, bbB, N);

  // cross attention (seq len 1): out_proj(v_proj(low)), then LN
  k_gemm_bt<256, 4><<<dim3(4, N / 16), 32, 0, stream>>>(bbB, 256, wvb, 256, att_bv, nullptr, bbA, 256);
  k_gemm_bt<256, 4><<<dim3(4, N / 16), 32, 0, stream>>>(bbA, 256, wob, 256, att_bo, bufB, nullptr, 256);
  k_layernorm<256><<<N / 8, w8, 0, stream>>>(bufB, cng, cnb, 0, nullptr, bufA, nullptr, N);

  // enc_in = x_t_hidden + pos_em + cross ; enc = LN((DAD@enc_in)@de_conv+b) + pro2
  k_encin<<<N, 256, 0, stream>>>(xth, bufA, ts, dinv, encin, Zt);
  k_dad_gemm<4><<<dim3(4, N / 16), 32, 0, stream>>>(Ab, Zt, dinv, encin, 256, Yb);
  k_gemm_bt<256, 4><<<dim3(1, N / 16), 32, 0, stream>>>(Yb, 256, dct, 256, de_bias, bufB, nullptr, 64);
  k_layernorm<64><<<N / 8, w8, 0, stream>>>(bufB, dng, dnb, 0, pro2, encf, nullptr, N);

  // self-loop masked attention GCN
  k_gemm_bt<64, 4><<<dim3(1, N / 16), 32, 0, stream>>>(featb, 64, wqt, 64, nullptr, nullptr, Qb, 64);
  k_gemm_bt<64, 4><<<dim3(1, N / 16), 32, 0, stream>>>(featb, 64, wkt, 64, nullptr, nullptr, Kbm, 64);
  k_gemm_bt<64, 4><<<dim3(1, N / 16), 32, 0, stream>>>(featb, 64, wvt, 64, nullptr, Vf, nullptr, 64);
  k_tconv<<<(N * F + 255) / 256, 256, 0, stream>>>(Vf, Vt, N, 64, 1.f);
  k_flash<<<dim3(N / 16, 4), 32, 0, stream>>>(Qb, Kbm, Vt, Mt, Ob);
  k_gemm_bt<64, 4><<<dim3(1, N / 16), 32, 0, stream>>>(Ob, 64, wot, 64, nullptr, enh, nullptr, 64);

  // final head
  k_final<<<N / 8, w8, 0, stream>>>(encf, enh, de_mlp_w, de_mlp_b, pro2, (float*)d_out, N);
}
